// SwinTransformerBlock_31748398252795
// MI455X (gfx1250) — matrix operand order repaired and checked by reading, not compiled, not run on hardware
//
#include <hip/hip_runtime.h>
#include <math.h>

// ---------------- problem constants ----------------
#define DIMC   256
#define HEADS  8
#define HD     32          // head dim
#define WSZ    7
#define SSZ    3
#define BATCH  16
#define HIMG   56
#define NTOK   49          // tokens per window
#define NWIN   64          // windows per image (8x8)
#define ROWS   50176       // BATCH*NWIN*NTOK == BATCH*HIMG*HIMG
#define HID    1024
#define QSCALE 0.17677669529663687f   // 1/sqrt(32)

typedef __attribute__((ext_vector_type(16))) _Float16 v16h;
typedef __attribute__((ext_vector_type(8)))  float    v8f;

// ---------------- CDNA5 async global->LDS helpers --------------------------
// Generic pointers to __shared__ are aperture|offset; addr[31:0] is the raw
// LDS byte address (ISA 10.2 aperture mapping), which is what VDST needs.
__device__ inline unsigned lds_addr_of(const void* p) {
  return (unsigned)(size_t)p;
}
// Two GLOBAL_LOAD_ASYNC_TO_LDS_B128 covering 32 contiguous bytes.
// INST_OFFSET advances BOTH the LDS and global addresses (ISA 08 §4.4),
// so one 32-bit VGPR offset serves both halves.
__device__ inline void async_copy32(unsigned ldsA, unsigned voff, const void* sbase) {
  asm volatile("global_load_async_to_lds_b128 %0, %1, %2\n\t"
               "global_load_async_to_lds_b128 %0, %1, %2 offset:16"
               :: "v"(ldsA), "v"(voff), "s"(sbase) : "memory");
}
__device__ inline void wait_async_le4() { asm volatile("s_wait_asynccnt 0x4" ::: "memory"); }
__device__ inline void wait_async_0()  { asm volatile("s_wait_asynccnt 0x0" ::: "memory"); }

// ---------------- WMMA fragment loaders (contiguous k = 16h + e for A and B, wave32) -------
// A 16x32 f16 fragment: lane L holds row M=L&15, h=L>>4.
//   element e : K = 16h + e (contiguous)
__device__ inline v16h load_frag_a(const _Float16* tile, int lane, int ld) {
  int m = lane & 15, h = lane >> 4;
  const _Float16* row = tile + m * ld + 16 * h;
  v16h r;
#pragma unroll
  for (int e = 0; e < 8; ++e)  r[e] = row[e];
#pragma unroll
  for (int e = 8; e < 16; ++e) r[e] = row[e];
  return r;
}
// B 32x16 f16 fragment from a transposed tile Bt[n][k]:
//   lane L holds col N=L&15, h=L>>4; element e: K = 16h + e (contiguous 32B)
__device__ inline v16h load_frag_b(const _Float16* tileT, int lane, int ld) {
  int n = lane & 15, h = lane >> 4;
  const _Float16* row = tileT + n * ld + 16 * h;
  v16h r;
#pragma unroll
  for (int e = 0; e < 16; ++e) r[e] = row[e];
  return r;
}

// ---------------- weight convert + transpose (f32 KxN -> f16 NxK) ----------
__global__ void wconv_kernel(const float* __restrict__ src, _Float16* __restrict__ dst,
                             int K, int N) {
  int i = blockIdx.x * blockDim.x + threadIdx.x;
  if (i < K * N) {
    int k = i / N, n = i - k * N;
    dst[(size_t)n * K + k] = (_Float16)src[i];
  }
}

// ------- fused LayerNorm (+optional cyclic shift & window partition) -> f16 -
__global__ __launch_bounds__(256)
void ln_win_kernel(const float* __restrict__ x, const float* __restrict__ g,
                   const float* __restrict__ b, _Float16* __restrict__ out, int shifted) {
  __shared__ float red[256];
  int row = blockIdx.x;
  int c   = threadIdx.x;
  int src = row;
  if (shifted) {
    int w = row / NTOK, t = row - w * NTOK;
    int bb = w >> 6, widx = w & 63;
    int wy = widx >> 3, wx = widx & 7;
    int ty = t / WSZ, tx = t - ty * WSZ;
    int y  = wy * WSZ + ty, xp = wx * WSZ + tx;
    int yy = (y + SSZ) % HIMG, xx = (xp + SSZ) % HIMG;   // undo roll(-SS)
    src = bb * (HIMG * HIMG) + yy * HIMG + xx;
  }
  float v = x[(size_t)src * DIMC + c];
  red[c] = v; __syncthreads();
#pragma unroll
  for (int s = 128; s > 0; s >>= 1) { if (c < s) red[c] += red[c + s]; __syncthreads(); }
  float mean = red[0] * (1.0f / DIMC);
  __syncthreads();
  float d = v - mean;
  red[c] = d * d; __syncthreads();
#pragma unroll
  for (int s = 128; s > 0; s >>= 1) { if (c < s) red[c] += red[c + s]; __syncthreads(); }
  float var = red[0] * (1.0f / DIMC);
  float y = d * rsqrtf(var + 1e-5f) * g[c] + b[c];
  out[(size_t)row * DIMC + c] = (_Float16)y;
}

// ---------------- generic WMMA GEMM: C(MxN) = A(MxK,f16) @ Bt(NxK,f16) ------
// 128x128 block tile, 8 waves (4x2), each wave 2x4 tiles of 16x16, BK=32.
// Double-buffered LDS staged with async global->LDS (4 b128 ops / thread).
// mode 0: outH = (acc+bias)*scale ; mode 1: outH = gelu_exact(acc+bias) ;
// mode 2: outF = acc+bias+res (final residual)
#define BMT 128
#define BNT 128
#define BKT 32
#define ALD 40
#define BLD 40
__global__ __launch_bounds__(256)
void gemm_wmma_kernel(const _Float16* __restrict__ A, const _Float16* __restrict__ Bt,
                      const float* __restrict__ bias, _Float16* __restrict__ outH,
                      float* __restrict__ outF, const float* __restrict__ res,
                      int M, int N, int K, int mode, float scale) {
  __shared__ _Float16 As[2][BMT * ALD];
  __shared__ _Float16 Bs[2][BNT * BLD];
  int tid  = threadIdx.x;
  int lane = tid & 31;
  int wv   = tid >> 5;        // 0..7
  int wrow = wv >> 1;         // 0..3 : 32-row strip
  int wcol = wv & 1;          // 0..1 : 64-col strip
  int bm = blockIdx.y * BMT;
  int bn = blockIdx.x * BNT;

  v8f acc[2][4];
#pragma unroll
  for (int i = 0; i < 2; ++i)
#pragma unroll
    for (int j = 0; j < 4; ++j) acc[i][j] = (v8f){0.f,0.f,0.f,0.f,0.f,0.f,0.f,0.f};

  int lrow = tid >> 1;            // 0..127
  int lseg = (tid & 1) * 16;      // half offset within 32-K chunk

  const void* Ablk = (const void*)(A  + (size_t)bm * K);   // uniform per block
  const void* Bblk = (const void*)(Bt + (size_t)bn * K);
  unsigned lA0 = lds_addr_of(&As[0][lrow * ALD + lseg]);
  unsigned lB0 = lds_addr_of(&Bs[0][lrow * BLD + lseg]);
  unsigned lA1 = lds_addr_of(&As[1][lrow * ALD + lseg]);
  unsigned lB1 = lds_addr_of(&Bs[1][lrow * BLD + lseg]);
  unsigned rowOffB = (unsigned)((lrow * K + lseg) * 2);    // bytes, same for A & B

  int nsteps = K / BKT;
  // prologue: stage tile 0 into buffer 0
  async_copy32(lA0, rowOffB, Ablk);
  async_copy32(lB0, rowOffB, Bblk);

  for (int s = 0; s < nsteps; ++s) {
    int cur = s & 1;
    if (s + 1 < nsteps) {
      unsigned off = rowOffB + (unsigned)((s + 1) * BKT * 2);
      async_copy32(cur ? lA0 : lA1, off, Ablk);
      async_copy32(cur ? lB0 : lB1, off, Bblk);
      wait_async_le4();           // previous tile's 4 ops retired
    } else {
      wait_async_0();
    }
    __syncthreads();

    const _Float16* Ac = As[cur];
    const _Float16* Bc = Bs[cur];
    v16h afrag[2], bfrag[4];
#pragma unroll
    for (int rt = 0; rt < 2; ++rt)
      afrag[rt] = load_frag_a(Ac + (wrow * 32 + rt * 16) * ALD, lane, ALD);
#pragma unroll
    for (int ct = 0; ct < 4; ++ct)
      bfrag[ct] = load_frag_b(Bc + (wcol * 64 + ct * 16) * BLD, lane, BLD);
#pragma unroll
    for (int rt = 0; rt < 2; ++rt)
#pragma unroll
      for (int ct = 0; ct < 4; ++ct)
        acc[rt][ct] = __builtin_amdgcn_wmma_f32_16x16x32_f16(
            false, afrag[rt], false, bfrag[ct], (short)0, acc[rt][ct], false, false);
    __syncthreads();              // all waves done reading before next overwrite
  }

  int nl = lane & 15, h = lane >> 4;
#pragma unroll
  for (int rt = 0; rt < 2; ++rt)
#pragma unroll
    for (int ct = 0; ct < 4; ++ct) {
      int col = bn + wcol * 64 + ct * 16 + nl;
      float bv = bias ? bias[col] : 0.0f;
#pragma unroll
      for (int r = 0; r < 8; ++r) {
        int row = bm + wrow * 32 + rt * 16 + r + 8 * h;
        float v = acc[rt][ct][r] + bv;
        size_t idx = (size_t)row * N + col;
        if (mode == 0) {
          outH[idx] = (_Float16)(v * scale);
        } else if (mode == 1) {
          outH[idx] = (_Float16)(0.5f * v * (1.0f + erff(v * 0.70710678118f)));
        } else {
          outF[idx] = v + res[idx];
        }
      }
    }
}

// ---------------- windowed attention: one block per (window, head) ---------
#define QLD 40
#define VLD 72
#define SLD 65
#define PLD 72
__global__ __launch_bounds__(128)
void attn_kernel(const _Float16* __restrict__ qf, const _Float16* __restrict__ kf,
                 const _Float16* __restrict__ vf, const float* __restrict__ rel_table,
                 const float* __restrict__ attn_mask, _Float16* __restrict__ outA) {
  __shared__ _Float16 Qs[64 * QLD];
  __shared__ _Float16 Ks[64 * QLD];
  __shared__ _Float16 Vt[32 * VLD];   // V transposed: Vt[d][m]
  __shared__ float    S[64 * SLD];
  __shared__ _Float16 P[64 * PLD];

  int w   = blockIdx.x;     // 0..1023
  int hh  = blockIdx.y;     // 0..7
  int tid = threadIdx.x, lane = tid & 31, wv = tid >> 5;

  size_t base = (size_t)w * NTOK * DIMC + hh * HD;

  // stage Q,K (64 rows x 32 halfs) with async global->LDS; zero-pad rows>=49
  {
    int r = tid >> 1, seg = (tid & 1) * 16;
    unsigned lq = lds_addr_of(Qs + r * QLD + seg);
    unsigned lk = lds_addr_of(Ks + r * QLD + seg);
    if (r < NTOK) {
      unsigned goff = (unsigned)((r * DIMC + seg) * 2);   // bytes
      async_copy32(lq, goff, (const void*)(qf + base));
      async_copy32(lk, goff, (const void*)(kf + base));
    } else {
      uint4 z; z.x = z.y = z.z = z.w = 0u;
      uint4* sq = (uint4*)(Qs + r * QLD + seg);
      uint4* sk = (uint4*)(Ks + r * QLD + seg);
      sq[0] = z; sq[1] = z; sk[0] = z; sk[1] = z;
    }
  }
  // load V transposed (threads 0..63 each handle one m-row)
  if (tid < 64) {
    int m = tid;
    const _Float16* gv = vf + base + (size_t)m * DIMC;
#pragma unroll
    for (int d = 0; d < HD; ++d)
      Vt[d * VLD + m] = (m < NTOK) ? gv[d] : (_Float16)0.0f;
  }
  wait_async_0();
  __syncthreads();

  // S = Q @ K^T  (4x4 tiles, wave wv owns one 16-row strip)
  {
    v16h a = load_frag_a(Qs + wv * 16 * QLD, lane, QLD);
    int nl = lane & 15, h = lane >> 4;
#pragma unroll
    for (int ct = 0; ct < 4; ++ct) {
      v16h b = load_frag_b(Ks + ct * 16 * QLD, lane, QLD);
      v8f acc = (v8f){0.f,0.f,0.f,0.f,0.f,0.f,0.f,0.f};
      acc = __builtin_amdgcn_wmma_f32_16x16x32_f16(false, a, false, b, (short)0, acc, false, false);
#pragma unroll
      for (int r = 0; r < 8; ++r)
        S[(wv * 16 + r + 8 * h) * SLD + ct * 16 + nl] = acc[r];
    }
  }
  __syncthreads();

  // bias + mask + softmax over valid 49 cols; write P (f16, padded w/ zeros)
  if (tid < NTOK) {
    int n = tid;
    int nty = n / WSZ, ntx = n - nty * WSZ;
    const float* mrow = attn_mask + ((size_t)(w & 63) * NTOK + n) * NTOK;
    float mx = -1e30f;
    for (int m = 0; m < NTOK; ++m) {
      int mty = m / WSZ, mtx = m - mty * WSZ;
      int rel = (nty - mty + WSZ - 1) * (2 * WSZ - 1) + (ntx - mtx + WSZ - 1);
      float v = S[n * SLD + m] + rel_table[rel * HEADS + hh] + mrow[m];
      S[n * SLD + m] = v;
      mx = fmaxf(mx, v);
    }
    float sum = 0.0f;
    for (int m = 0; m < NTOK; ++m) {
      float e = __expf(S[n * SLD + m] - mx);
      S[n * SLD + m] = e;
      sum += e;
    }
    float rinv = 1.0f / sum;
    for (int m = 0; m < NTOK; ++m) P[n * PLD + m] = (_Float16)(S[n * SLD + m] * rinv);
    for (int m = NTOK; m < 64; ++m) P[n * PLD + m] = (_Float16)0.0f;
  } else if (tid < 64) {
    for (int m = 0; m < 64; ++m) P[tid * PLD + m] = (_Float16)0.0f;
  }
  __syncthreads();

  // O = P @ V  (64x32, K=64 -> 2 WMMA per tile)
  {
    int nl = lane & 15, h = lane >> 4;
#pragma unroll
    for (int ct = 0; ct < 2; ++ct) {
      v8f acc = (v8f){0.f,0.f,0.f,0.f,0.f,0.f,0.f,0.f};
#pragma unroll
      for (int kc = 0; kc < 2; ++kc) {
        v16h a = load_frag_a(P  + wv * 16 * PLD + kc * 32, lane, PLD);
        v16h b = load_frag_b(Vt + ct * 16 * VLD + kc * 32, lane, VLD);
        acc = __builtin_amdgcn_wmma_f32_16x16x32_f16(false, a, false, b, (short)0, acc, false, false);
      }
#pragma unroll
      for (int r = 0; r < 8; ++r) {
        int row = wv * 16 + r + 8 * h;
        if (row < NTOK)
          outA[(size_t)(w * NTOK + row) * DIMC + hh * HD + ct * 16 + nl] = (_Float16)acc[r];
      }
    }
  }
}

// ------- reverse shift + window-reverse + residual (scatter, bijective) ----
__global__ __launch_bounds__(256)
void rev_res_kernel(const _Float16* __restrict__ projout, const float* __restrict__ x1,
                    float* __restrict__ xres) {
  int row = blockIdx.x;             // window-token row of projout
  int c   = threadIdx.x;
  int w = row / NTOK, t = row - w * NTOK;
  int bb = w >> 6, widx = w & 63;
  int wy = widx >> 3, wx = widx & 7;
  int ty = t / WSZ, tx = t - ty * WSZ;
  int y  = wy * WSZ + ty, xp = wx * WSZ + tx;
  int yy = (y + SSZ) % HIMG, xx = (xp + SSZ) % HIMG;   // roll(+SS)
  size_t dst = (size_t)(bb * (HIMG * HIMG) + yy * HIMG + xx) * DIMC + c;
  xres[dst] = x1[dst] + (float)projout[(size_t)row * DIMC + c];
}

// --------------------------------------------------------------------------
extern "C" void kernel_launch(void* const* d_in, const int* in_sizes, int n_in,
                              void* d_out, int out_size, void* d_ws, size_t ws_size,
                              hipStream_t stream) {
  (void)in_sizes; (void)n_in; (void)out_size; (void)ws_size;
  const float* x1        = (const float*)d_in[0];
  const float* x2        = (const float*)d_in[1];
  const float* x3        = (const float*)d_in[2];
  const float* attn_mask = (const float*)d_in[3];
  const float* g1        = (const float*)d_in[4];
  const float* bb1       = (const float*)d_in[5];
  const float* wq        = (const float*)d_in[6];
  const float* bq        = (const float*)d_in[7];
  const float* wk        = (const float*)d_in[8];
  const float* bk        = (const float*)d_in[9];
  const float* wvw       = (const float*)d_in[10];
  const float* bv        = (const float*)d_in[11];
  const float* rel_table = (const float*)d_in[12];
  const float* wp        = (const float*)d_in[13];
  const float* bp        = (const float*)d_in[14];
  const float* g2        = (const float*)d_in[15];
  const float* bb2       = (const float*)d_in[16];
  const float* fw1       = (const float*)d_in[17];
  const float* fb1       = (const float*)d_in[18];
  const float* fw2       = (const float*)d_in[19];
  const float* fb2       = (const float*)d_in[20];
  float* out = (float*)d_out;

  // ---- workspace layout (bytes), with temporal overlays ----
  char* ws = (char*)d_ws;
  const size_t F16ROWxC = (size_t)ROWS * DIMC * sizeof(_Float16);   // 25,690,112
  size_t off = 0;
  auto take = [&](size_t bytes) { size_t o = off; off = (off + bytes + 255) & ~(size_t)255; return o; };
  size_t oWq  = take(DIMC * DIMC * 2);
  size_t oWk  = take(DIMC * DIMC * 2);
  size_t oWv  = take(DIMC * DIMC * 2);
  size_t oWp  = take(DIMC * DIMC * 2);
  size_t oF1  = take((size_t)DIMC * HID * 2);
  size_t oF2  = take((size_t)HID * DIMC * 2);
  size_t oRA  = take((size_t)ROWS * HID * 2);   // Xw1..3 -> projout -> hidden
  size_t oRB  = take(3 * F16ROWxC);             // q,k,v
  size_t oRC  = take(F16ROWxC);                 // attnout -> h1in
  size_t oRD  = take((size_t)ROWS * DIMC * 4);  // xres f32

  _Float16* wqT  = (_Float16*)(ws + oWq);
  _Float16* wkT  = (_Float16*)(ws + oWk);
  _Float16* wvT  = (_Float16*)(ws + oWv);
  _Float16* wpT  = (_Float16*)(ws + oWp);
  _Float16* fw1T = (_Float16*)(ws + oF1);
  _Float16* fw2T = (_Float16*)(ws + oF2);
  _Float16* Xw1  = (_Float16*)(ws + oRA);
  _Float16* Xw2  = Xw1 + (size_t)ROWS * DIMC;
  _Float16* Xw3  = Xw2 + (size_t)ROWS * DIMC;
  _Float16* projout = (_Float16*)(ws + oRA);          // overlay (Xw dead)
  _Float16* hidden  = (_Float16*)(ws + oRA);          // overlay (projout dead)
  _Float16* qf   = (_Float16*)(ws + oRB);
  _Float16* kf   = qf + (size_t)ROWS * DIMC;
  _Float16* vf   = kf + (size_t)ROWS * DIMC;
  _Float16* attnout = (_Float16*)(ws + oRC);
  _Float16* h1in    = (_Float16*)(ws + oRC);          // overlay (attnout dead)
  float*    xres    = (float*)(ws + oRD);

  // ---- 1) weight convert + transpose to f16 NxK ----
  wconv_kernel<<<(DIMC*DIMC + 255)/256, 256, 0, stream>>>(wq,  wqT,  DIMC, DIMC);
  wconv_kernel<<<(DIMC*DIMC + 255)/256, 256, 0, stream>>>(wk,  wkT,  DIMC, DIMC);
  wconv_kernel<<<(DIMC*DIMC + 255)/256, 256, 0, stream>>>(wvw, wvT,  DIMC, DIMC);
  wconv_kernel<<<(DIMC*DIMC + 255)/256, 256, 0, stream>>>(wp,  wpT,  DIMC, DIMC);
  wconv_kernel<<<(DIMC*HID  + 255)/256, 256, 0, stream>>>(fw1, fw1T, DIMC, HID);
  wconv_kernel<<<(HID*DIMC  + 255)/256, 256, 0, stream>>>(fw2, fw2T, HID,  DIMC);

  // ---- 2) LN1 + cyclic shift + window partition (f16) ----
  ln_win_kernel<<<ROWS, 256, 0, stream>>>(x1, g1, bb1, Xw1, 1);
  ln_win_kernel<<<ROWS, 256, 0, stream>>>(x2, g1, bb1, Xw2, 1);
  ln_win_kernel<<<ROWS, 256, 0, stream>>>(x3, g1, bb1, Xw3, 1);

  // ---- 3) QKV projections (WMMA GEMM), q pre-scaled by hd^-0.5 ----
  dim3 gQKV(DIMC / BNT, ROWS / BMT);     // (2, 392)
  gemm_wmma_kernel<<<gQKV, 256, 0, stream>>>(Xw1, wqT, bq, qf, nullptr, nullptr,
                                             ROWS, DIMC, DIMC, 0, QSCALE);
  gemm_wmma_kernel<<<gQKV, 256, 0, stream>>>(Xw2, wkT, bk, kf, nullptr, nullptr,
                                             ROWS, DIMC, DIMC, 0, 1.0f);
  gemm_wmma_kernel<<<gQKV, 256, 0, stream>>>(Xw3, wvT, bv, vf, nullptr, nullptr,
                                             ROWS, DIMC, DIMC, 0, 1.0f);

  // ---- 4) windowed attention (bias + mask + softmax + PV) ----
  dim3 gAtt(BATCH * NWIN, HEADS);        // (1024, 8)
  attn_kernel<<<gAtt, 128, 0, stream>>>(qf, kf, vf, rel_table, attn_mask, attnout);

  // ---- 5) output projection ----
  gemm_wmma_kernel<<<gQKV, 256, 0, stream>>>(attnout, wpT, bp, projout, nullptr, nullptr,
                                             ROWS, DIMC, DIMC, 0, 1.0f);

  // ---- 6) reverse shift + window reverse + residual -> xres (f32) ----
  rev_res_kernel<<<ROWS, 256, 0, stream>>>(projout, x1, xres);

  // ---- 7) LN2 -> f16 ----
  ln_win_kernel<<<ROWS, 256, 0, stream>>>(xres, g2, bb2, h1in, 0);

  // ---- 8) MLP fc1 + exact GELU ----
  dim3 gF1(HID / BNT, ROWS / BMT);       // (8, 392)
  gemm_wmma_kernel<<<gF1, 256, 0, stream>>>(h1in, fw1T, fb1, hidden, nullptr, nullptr,
                                            ROWS, HID, DIMC, 1, 1.0f);

  // ---- 9) MLP fc2 + residual -> d_out (f32) ----
  dim3 gF2(DIMC / BNT, ROWS / BMT);      // (2, 392)
  gemm_wmma_kernel<<<gF2, 256, 0, stream>>>(hidden, fw2T, fb2, nullptr, out, xres,
                                            ROWS, DIMC, HID, 2, 1.0f);
}
